// SemiAspectSent_27754078667141
// MI455X (gfx1250) — matrix operand order repaired and checked by reading, not compiled, not run on hardware
//
#include <hip/hip_runtime.h>

// Problem constants (from reference setup_inputs)
#define B_   64
#define L_   512
#define H_   256
#define HD_  128
#define G_   512   // 4*HD_

typedef __attribute__((ext_vector_type(16))) _Float16 v16h;
typedef __attribute__((ext_vector_type(8)))  _Float16 v8h;
typedef __attribute__((ext_vector_type(8)))  float    v8f;
typedef __attribute__((ext_vector_type(4)))  float    v4f;

// Fast transcendentals: v_rcp_f32 / v_exp_f32 based (gate pre-activations are
// O(1) here: f16 inputs, 0.05-scaled weights — no range issues).
__device__ __forceinline__ float fast_rcp(float x) { return __builtin_amdgcn_rcpf(x); }
__device__ __forceinline__ float sigm(float x)     { return fast_rcp(1.0f + __expf(-x)); }
__device__ __forceinline__ float fast_tanh(float x) {
  // tanh(x) = 1 - 2/(exp(2x)+1)
  return __builtin_fmaf(-2.0f, fast_rcp(__expf(2.0f * x) + 1.0f), 1.0f);
}

// Concatenate two 8-half loads into a 16-half WMMA operand.
__device__ __forceinline__ v16h ld16(const _Float16* p0, const _Float16* p1) {
  return __builtin_shufflevector(*(const v8h*)p0, *(const v8h*)p1,
                                 0,1,2,3,4,5,6,7,8,9,10,11,12,13,14,15);
}

__global__ void cvt_f32_to_f16(const float* __restrict__ in, _Float16* __restrict__ out, int n) {
  int i = blockIdx.x * blockDim.x + threadIdx.x;
  if (i < n) out[i] = (_Float16)in[i];
}

__global__ void zero_out_f32(float* __restrict__ p, int n4) {
  int i = blockIdx.x * blockDim.x + threadIdx.x;
  if (i < n4) ((v4f*)p)[i] = (v4f){0.f, 0.f, 0.f, 0.f};
}

// One workgroup (8 wave32) processes 32 consecutive sequences r = b*L + i.
// Wave w owns hidden slice [16w, 16w+16) for all 4 gates, so i/f/g/o for a
// hidden unit land in the same lane/slot of four f32 accumulators.
__global__ __launch_bounds__(256, 1) void lstm_span_wmma(
    const _Float16* __restrict__ feats16,  // [B][L][H] f16
    const _Float16* __restrict__ Wih16,    // [G][H]  f16 (row-major, = W_ih)
    const _Float16* __restrict__ Whh16,    // [G][HD] f16
    const float* __restrict__ b_ih, const float* __restrict__ b_hh,
    const float* __restrict__ W_tri, const float* __restrict__ b_tri,
    const int* __restrict__ lens, float* __restrict__ out)
{
  __shared__ __align__(16) _Float16 Xs[36][264];        // feats rows i0..i0+35 (zero past L)
  __shared__ __align__(16) _Float16 Hall[4][32][136];   // h after step t, per row, f16

  const int tid  = threadIdx.x;
  const int lane = tid & 31;
  const int wave = tid >> 5;
  const int r0   = blockIdx.x * 32;
  const int b    = r0 / L_;
  const int i0   = r0 % L_;   // 32 | L_, all rows share b

  // ---- cooperative staging of X into LDS (v8h = 16B chunks) ----
  for (int idx = tid; idx < 36 * (H_ / 8); idx += 256) {
    const int row = idx >> 5;
    const int k8  = idx & 31;
    const int l   = i0 + row;
    v8h v = {};
    if (l < L_) v = *(const v8h*)(feats16 + ((size_t)b * L_ + l) * H_ + k8 * 8);
    *(v8h*)&Xs[row][k8 * 8] = v;
  }
  __syncthreads();

  const int kw       = wave << 4;          // hidden-slice base for this wave
  const int ncol     = kw + (lane & 15);   // hidden index this lane owns in C tiles
  const int hi16     = lane >> 4;          // half-wave selector
  const int mrow_off = hi16 << 3;          // C layout: M = vgpr + 8*(lane>=16)

  // biases are step-invariant; gate order i,f,g,o in chunks of HD_
  const float bI = b_ih[0 * HD_ + ncol] + b_hh[0 * HD_ + ncol];
  const float bF = b_ih[1 * HD_ + ncol] + b_hh[1 * HD_ + ncol];
  const float bG = b_ih[2 * HD_ + ncol] + b_hh[2 * HD_ + ncol];
  const float bO = b_ih[3 * HD_ + ncol] + b_hh[3 * HD_ + ncol];

  const v8f z8 = {0.f, 0.f, 0.f, 0.f, 0.f, 0.f, 0.f, 0.f};
  v8f cst0 = z8, cst1 = z8;   // LSTM cell state, register-resident across steps

  for (int t = 0; t < 4; ++t) {
    v8f acc[4][2];
    #pragma unroll
    for (int q = 0; q < 4; ++q) { acc[q][0] = z8; acc[q][1] = z8; }

    // ---- input projection: gates += X_t @ W_ih^T   (K = 256, 8 chunks of 32) ----
    #pragma unroll
    for (int kc = 0; kc < 8; ++kc) {
      const int ka = kc * 32 + (hi16 ? 8 : 0);   // A layout K sub-offset
      const _Float16* pa0 = &Xs[(lane & 15) + t][ka];
      const _Float16* pa1 = &Xs[16 + (lane & 15) + t][ka];
      const v16h a0 = ld16(pa0, pa0 + 16);
      const v16h a1 = ld16(pa1, pa1 + 16);
      #pragma unroll
      for (int q = 0; q < 4; ++q) {
        const _Float16* wp = Wih16 + (size_t)(q * HD_ + ncol) * H_ + kc * 32 + (hi16 ? 8 : 0);
        const v16h bm = ld16(wp, wp + 16);
        acc[q][0] = __builtin_amdgcn_wmma_f32_16x16x32_f16(false, a0, false, bm, (short)0, acc[q][0], false, false);
        acc[q][1] = __builtin_amdgcn_wmma_f32_16x16x32_f16(false, a1, false, bm, (short)0, acc[q][1], false, false);
      }
    }

    // ---- recurrent projection: gates += h_{t-1} @ W_hh^T  (K = 128, 4 chunks) ----
    if (t > 0) {
      #pragma unroll
      for (int kc = 0; kc < 4; ++kc) {
        const int ka = kc * 32 + (hi16 ? 8 : 0);
        const _Float16* pa0 = &Hall[t - 1][(lane & 15)][ka];
        const _Float16* pa1 = &Hall[t - 1][16 + (lane & 15)][ka];
        const v16h a0 = ld16(pa0, pa0 + 16);
        const v16h a1 = ld16(pa1, pa1 + 16);
        #pragma unroll
        for (int q = 0; q < 4; ++q) {
          const _Float16* wp = Whh16 + (size_t)(q * HD_ + ncol) * HD_ + kc * 32 + (hi16 ? 8 : 0);
          const v16h bm = ld16(wp, wp + 16);
          acc[q][0] = __builtin_amdgcn_wmma_f32_16x16x32_f16(false, a0, false, bm, (short)0, acc[q][0], false, false);
          acc[q][1] = __builtin_amdgcn_wmma_f32_16x16x32_f16(false, a1, false, bm, (short)0, acc[q][1], false, false);
        }
      }
    }

    // ---- LSTM nonlinearity (i,f,g,o live in the same lane/slot across acc[q]) ----
    #pragma unroll
    for (int j = 0; j < 8; ++j) {
      {
        const float iv = acc[0][0][j] + bI;
        const float fv = acc[1][0][j] + bF;
        const float gv = acc[2][0][j] + bG;
        const float ov = acc[3][0][j] + bO;
        const float c  = sigm(fv) * cst0[j] + sigm(iv) * fast_tanh(gv);
        cst0[j] = c;
        Hall[t][j + mrow_off][ncol] = (_Float16)(sigm(ov) * fast_tanh(c));
      }
      {
        const float iv = acc[0][1][j] + bI;
        const float fv = acc[1][1][j] + bF;
        const float gv = acc[2][1][j] + bG;
        const float ov = acc[3][1][j] + bO;
        const float c  = sigm(fv) * cst1[j] + sigm(iv) * fast_tanh(gv);
        cst1[j] = c;
        Hall[t][16 + j + mrow_off][ncol] = (_Float16)(sigm(ov) * fast_tanh(c));
      }
    }
    __syncthreads();   // publish Hall[t] for next step / emit
  }

  // ---- emit: (h_span @ W_tri^T + b_tri) * masks, scattered into the band ----
  // tid -> (row m_local, width w, output channel e): 32 * 4 * 2 = 256
  const int m_local = tid >> 3;
  const int w       = ((tid >> 1) & 3) + 1;
  const int e       = tid & 1;
  const int i       = i0 + m_local;
  const int rem     = lens[b] - i;
  int total = rem < 0 ? 0 : (rem > w ? w : rem);
  const int Lsel = total > 0 ? total : 1;

  float s = 0.f;
  const float* wt = W_tri + e * HD_;
  #pragma unroll
  for (int k8 = 0; k8 < HD_ / 8; ++k8) {
    const v8h hv = *(const v8h*)&Hall[Lsel - 1][m_local][k8 * 8];   // ds_load_b128
    #pragma unroll
    for (int u = 0; u < 8; ++u) s += (float)hv[u] * wt[k8 * 8 + u];
  }
  s += b_tri[e];
  if (total <= 0) s = 0.f;   // row_mask

  const int j = i + w - 1;
  if (j < L_)                 // valid_span; clamped-j contributions are zero -> skip
    out[(((size_t)b * L_ + i) * L_ + j) * 2 + e] = s;
}

extern "C" void kernel_launch(void* const* d_in, const int* in_sizes, int n_in,
                              void* d_out, int out_size, void* d_ws, size_t ws_size,
                              hipStream_t stream) {
  const float* feats = (const float*)d_in[0];   // [64][512][256]
  const float* W_ih  = (const float*)d_in[1];   // [512][256]
  const float* W_hh  = (const float*)d_in[2];   // [512][128]
  const float* b_ih  = (const float*)d_in[3];
  const float* b_hh  = (const float*)d_in[4];
  const float* W_tri = (const float*)d_in[5];   // [2][128]
  const float* b_tri = (const float*)d_in[6];
  const int*   lens  = (const int*)d_in[7];
  float* out = (float*)d_out;

  // workspace: f16 copies (feats 16.8MB + W_ih 256KB + W_hh 128KB)
  _Float16* feats16 = (_Float16*)d_ws;
  const size_t nF = (size_t)B_ * L_ * H_;       // 8,388,608
  _Float16* Wih16 = feats16 + nF;
  _Float16* Whh16 = Wih16 + (size_t)G_ * H_;

  cvt_f32_to_f16<<<(int)((nF + 255) / 256), 256, 0, stream>>>(feats, feats16, (int)nF);
  cvt_f32_to_f16<<<(G_ * H_ + 255) / 256, 256, 0, stream>>>(W_ih, Wih16, G_ * H_);
  cvt_f32_to_f16<<<(G_ * HD_ + 255) / 256, 256, 0, stream>>>(W_hh, Whh16, G_ * HD_);

  const int n4 = out_size / 4;                   // 128MB output zero-fill, b128 stores
  zero_out_f32<<<(n4 + 255) / 256, 256, 0, stream>>>(out, n4);

  lstm_span_wmma<<<(B_ * L_) / 32, 256, 0, stream>>>(
      feats16, Wih16, Whh16, b_ih, b_hh, W_tri, b_tri, lens, out);
}
